// VAEFidelityDiffusion_20005957665148
// MI455X (gfx1250) — matrix operand order repaired and checked by reading, not compiled, not run on hardware
//
#include <hip/hip_runtime.h>
#include <hip/hip_bf16.h>
#include <math.h>

typedef __bf16 bf16;
typedef __attribute__((ext_vector_type(16))) __bf16 bf16x16;
typedef __attribute__((ext_vector_type(8)))  float   f32x8;

#define Bsz  131072
#define Dd   64
#define Hh   512
#define ROWS 32      // rows per block
#define NT   256     // threads per block (8 waves, wave32)
#define SA   520     // activation LDS row stride in bf16 elems (512 + pad)
#define WS   40      // weight-tile LDS per-column stride in bf16 (32 + 8 pad)

union FragU { uint4 u[2]; bf16x16 v; };

__device__ __forceinline__ uint32_t lds_addr32(const void* p) {
  return (uint32_t)(uintptr_t)p;   // generic LDS addr: low 32 bits = LDS offset
}

#define WAIT_ASYNC() asm volatile("s_wait_asynccnt 0x0" ::: "memory")

// ---------------------------------------------------------------------------
// Async-stage a 32-K slab of a fragment-packed bf16 weight matrix into LDS.
// Global layout: [kslab][col][32] (each column's 32 K-values contiguous).
// LDS layout:    [col][WS] padded column stride -> bank-friendly b128 reads.
// Uses GLOBAL_LOAD_ASYNC_TO_LDS_B128 (ASYNCcnt) so the copy overlaps compute.
// ---------------------------------------------------------------------------
__device__ __forceinline__ void stage_async(const bf16* __restrict__ Wg, bf16* wt,
                                            int kk, int ncols, int tid) {
  const char* src = (const char*)(Wg + (size_t)kk * 32 * ncols);
  const int nvec = ncols * 4;            // 4 x 16B per 32-bf16 column
  for (int i = tid; i < nvec; i += NT) {
    int col = i >> 2, q = i & 3;
    uint32_t lds = lds_addr32(wt) + (uint32_t)(col * 5 + q) * 16u;
    uint32_t gof = (uint32_t)i * 16u;
    asm volatile("global_load_async_to_lds_b128 %0, %1, %2"
                 :: "v"(lds), "v"(gof), "s"(src)
                 : "memory");
  }
}

// A fragment: two aligned 16B LDS loads (K ka..ka+7, K 16+ka..16+ka+7)
__device__ __forceinline__ bf16x16 load_a(const bf16* ap) {
  FragU f;
  f.u[0] = *(const uint4*)(ap);
  f.u[1] = *(const uint4*)(ap + 16);
  return f.v;
}

// B fragment: two aligned 16B loads (K kb..kb+7, K 16+kb..16+kb+7)
__device__ __forceinline__ bf16x16 load_b(const bf16* bp) {
  FragU f;
  f.u[0] = *(const uint4*)(bp);
  f.u[1] = *(const uint4*)(bp + 16);
  return f.v;
}

// ---------------------------------------------------------------------------
// Fused GEMM (32 x K @ K x 512) + bias + LayerNorm + SiLU, result written back
// to the activation LDS buffer as bf16.  K = KT*32.  Double-buffered async
// weight staging; batched B-fragment loads feeding a WMMA chain.
// Wave w: row-group (w>>2)*16, col-group (w&3)*128 -> 8 C tiles of 16x16.
// ---------------------------------------------------------------------------
__device__ __forceinline__ void gemm512_ln_silu(
    bf16* act, bf16* wt0, bf16* wt1,
    const bf16* __restrict__ Wg, const float* __restrict__ bias,
    const float* __restrict__ gamma, const float* __restrict__ beta,
    float* rowsum, float* rowsq, float* meanv, float* rstdv,
    int KT, int tid)
{
  const int lane  = tid & 31;
  const int wave  = tid >> 5;
  const int rbase = (wave >> 2) * 16;
  const int cbase = (wave & 3) * 128;
  const int lr    = lane & 15;
  const int hi    = lane >> 4;   // 0: lanes 0-15, 1: lanes 16-31
  const int ka    = hi * 8;      // A-fragment K offset (16-bit A layout)
  const int kb    = hi * 8;      // B-fragment K offset (16-bit B layout)

  f32x8 acc[8] = {};
  stage_async(Wg, wt0, 0, 512, tid);
  for (int kk = 0; kk < KT; ++kk) {
    WAIT_ASYNC();
    __syncthreads();
    bf16* wt = (kk & 1) ? wt1 : wt0;
    if (kk + 1 < KT) stage_async(Wg, (kk & 1) ? wt0 : wt1, kk + 1, 512, tid);

    bf16x16 a = load_a(act + (rbase + lr) * SA + kk * 32 + ka);
    bf16x16 bfrag[8];
#pragma unroll
    for (int j = 0; j < 8; ++j)
      bfrag[j] = load_b(wt + (cbase + j * 16 + lr) * WS + kb);
#pragma unroll
    for (int j = 0; j < 8; ++j)
      acc[j] = __builtin_amdgcn_wmma_f32_16x16x32_bf16(
          false, a, false, bfrag[j], (short)0, acc[j], false, false);
  }
  __syncthreads();
  if (tid < ROWS) { rowsum[tid] = 0.f; rowsq[tid] = 0.f; }
  __syncthreads();

  // bias + per-lane LN partials from C fragments
  float s1[8], s2[8];
#pragma unroll
  for (int g = 0; g < 8; ++g) { s1[g] = 0.f; s2[g] = 0.f; }
#pragma unroll
  for (int j = 0; j < 8; ++j) {
    float bv = bias[cbase + j * 16 + lr];
#pragma unroll
    for (int g = 0; g < 8; ++g) {
      float v = acc[j][g] + bv;
      acc[j][g] = v;
      s1[g] += v; s2[g] += v * v;
    }
  }
  // butterfly over each 16-lane half (cols), then LDS add across col-groups
#pragma unroll
  for (int m = 1; m < 16; m <<= 1) {
#pragma unroll
    for (int g = 0; g < 8; ++g) {
      s1[g] += __shfl_xor(s1[g], m, 32);
      s2[g] += __shfl_xor(s2[g], m, 32);
    }
  }
  if (lr == 0) {
#pragma unroll
    for (int g = 0; g < 8; ++g) {
      int r = rbase + hi * 8 + g;
      atomicAdd(&rowsum[r], s1[g]);
      atomicAdd(&rowsq[r],  s2[g]);
    }
  }
  __syncthreads();
  if (tid < ROWS) {
    float m = rowsum[tid] * (1.0f / 512.0f);
    float v = rowsq[tid] * (1.0f / 512.0f) - m * m;
    meanv[tid] = m;
    rstdv[tid] = rsqrtf(v + 1e-5f);
  }
  __syncthreads();
#pragma unroll
  for (int j = 0; j < 8; ++j) {
    int col = cbase + j * 16 + lr;
    float gm = gamma[col], bt = beta[col];
#pragma unroll
    for (int g = 0; g < 8; ++g) {
      int r = rbase + hi * 8 + g;
      float v = (acc[j][g] - meanv[r]) * rstdv[r] * gm + bt;
      v = v / (1.0f + __expf(-v));           // SiLU
      act[r * SA + col] = (bf16)v;
    }
  }
  __syncthreads();
}

// ---------------------------------------------------------------------------
// GEMM (32 x 512 @ 512 x 64) + bias -> f32 pred buffer in LDS.
// Wave w: rows (w>>2)*16, cols (w&3)*16 (one 16x16 C tile per wave).
// ---------------------------------------------------------------------------
__device__ __forceinline__ void gemm64_out(
    bf16* act, bf16* wt0, bf16* wt1,
    const bf16* __restrict__ Wg, const float* __restrict__ bias,
    float* pred, int tid)
{
  const int lane  = tid & 31;
  const int wave  = tid >> 5;
  const int rbase = (wave >> 2) * 16;
  const int cbase = (wave & 3) * 16;
  const int lr    = lane & 15;
  const int hi    = lane >> 4;
  const int ka    = hi * 8;
  const int kb    = hi * 8;

  f32x8 acc = {};
  stage_async(Wg, wt0, 0, 64, tid);
  for (int kk = 0; kk < 16; ++kk) {
    WAIT_ASYNC();
    __syncthreads();
    bf16* wt = (kk & 1) ? wt1 : wt0;
    if (kk + 1 < 16) stage_async(Wg, (kk & 1) ? wt0 : wt1, kk + 1, 64, tid);

    bf16x16 a = load_a(act + (rbase + lr) * SA + kk * 32 + ka);
    bf16x16 b = load_b(wt + (cbase + lr) * WS + kb);
    acc = __builtin_amdgcn_wmma_f32_16x16x32_bf16(
        false, a, false, b, (short)0, acc, false, false);
  }
  __syncthreads();
  int col = cbase + lr;
  float bv = bias[col];
#pragma unroll
  for (int g = 0; g < 8; ++g) {
    int r = rbase + hi * 8 + g;
    pred[r * 64 + col] = acc[g] + bv;
  }
  __syncthreads();
}

// ---------------------------------------------------------------------------
// Utility kernels
// ---------------------------------------------------------------------------
__global__ void k_zero(float* s) {
  if (threadIdx.x == 0)
    for (int i = 0; i < 8; ++i) s[i] = 0.f;
}

// fp32 row-major [K][N] -> bf16 fragment-packed [kslab][col][32], K padded.
__global__ void k_cvt_pack(const float* __restrict__ src, bf16* __restrict__ dst,
                           int K, int Kpad, int N) {
  int i = blockIdx.x * blockDim.x + threadIdx.x;
  if (i < Kpad * N) {
    int t = i & 31;
    int rest = i >> 5;
    int col = rest % N;
    int kk = rest / N;
    int kg = kk * 32 + t;
    dst[i] = (kg < K) ? (bf16)src[(size_t)kg * N + col] : (bf16)0.0f;
  }
}

// ---------------------------------------------------------------------------
// Fused scorer: emb -> silu(LN(.@dW1+b)) -> silu(LN(.@dW2+b)) -> sigmoid(.@dW3+b)
// plus per-row radii and the three global reduction sums.
// ---------------------------------------------------------------------------
__global__ __launch_bounds__(NT) void k_scorer(
    const float* __restrict__ emb,
    const bf16* __restrict__ dW1b, const float* __restrict__ db1,
    const float* __restrict__ dg1, const float* __restrict__ dbt1,
    const bf16* __restrict__ dW2b, const float* __restrict__ db2,
    const float* __restrict__ dg2, const float* __restrict__ dbt2,
    const float* __restrict__ dW3, const float* __restrict__ db3,
    float* __restrict__ ws_scores, float* __restrict__ ws_radii,
    float* __restrict__ ws_scal)
{
  __shared__ __align__(16) bf16 act[ROWS * SA];
  __shared__ __align__(16) bf16 wt0[512 * WS];
  __shared__ __align__(16) bf16 wt1[512 * WS];
  __shared__ float rowsum[ROWS], rowsq[ROWS], meanv[ROWS], rstdv[ROWS];
  __shared__ float red[NT * 5];
  __shared__ float radL[ROWS], totL[ROWS];

  const int tid = threadIdx.x;
  const size_t base = (size_t)blockIdx.x * ROWS;

  // embeddings -> bf16 activations
  for (int i = tid; i < ROWS * Dd; i += NT) {
    int r = i >> 6, c = i & 63;
    act[r * SA + c] = (bf16)emb[(base + r) * Dd + c];
  }
  // per-row |x|^2 -> radii
  {
    int row = tid >> 3, seg = tid & 7;
    float p = 0.f;
    for (int c = seg * 8; c < seg * 8 + 8; ++c) {
      float e = emb[(base + row) * Dd + c];
      p += e * e;
    }
    red[tid] = p;
  }
  __syncthreads();
  if ((tid & 7) == 0) {
    int row = tid >> 3;
    float nx = 0.f;
    for (int s = 0; s < 8; ++s) nx += red[row * 8 + s];
    float arg = 1.0f + 2.0f * nx / (1.0f - nx + 1e-12f);
    float rad = acoshf(fmaxf(arg, 1.0f));
    radL[row] = rad;
    ws_radii[base + row] = rad;
  }
  __syncthreads();

  gemm512_ln_silu(act, wt0, wt1, dW1b, db1, dg1, dbt1, rowsum, rowsq, meanv, rstdv, 2, tid);
  gemm512_ln_silu(act, wt0, wt1, dW2b, db2, dg2, dbt2, rowsum, rowsq, meanv, rstdv, 16, tid);

  // head: 512 -> 5, sigmoid (vectorized LDS reads)
  {
    int row = tid >> 3, seg = tid & 7;
    float s[5] = {0.f, 0.f, 0.f, 0.f, 0.f};
    for (int kq = seg * 64; kq < seg * 64 + 64; kq += 8) {
      union { uint4 u; bf16 h[8]; } w;
      w.u = *(const uint4*)(act + row * SA + kq);
#pragma unroll
      for (int t = 0; t < 8; ++t) {
        float hv = (float)w.h[t];
#pragma unroll
        for (int c = 0; c < 5; ++c) s[c] += hv * dW3[(kq + t) * 5 + c];
      }
    }
#pragma unroll
    for (int c = 0; c < 5; ++c) red[tid * 5 + c] = s[c];
  }
  __syncthreads();
  if ((tid & 7) == 0) {
    int row = tid >> 3;
    float tot = 0.f;
#pragma unroll
    for (int c = 0; c < 5; ++c) {
      float x = db3[c];
      for (int s = 0; s < 8; ++s) x += red[(row * 8 + s) * 5 + c];
      float sc = 1.0f / (1.0f + __expf(-x));
      ws_scores[(base + row) * 5 + c] = sc;
      tot += sc;
    }
    totL[row] = tot * 0.2f;   // mean over 5
  }
  __syncthreads();
  if (tid == 0) {
    float st = 0.f, sr = 0.f, sr2 = 0.f;
    for (int r = 0; r < ROWS; ++r) {
      st += totL[r];
      float rd = radL[r];
      sr += rd; sr2 += rd * rd;
    }
    atomicAdd(&ws_scal[0], st);
    atomicAdd(&ws_scal[1], sr);
    atomicAdd(&ws_scal[2], sr2);
  }
}

// ---------------------------------------------------------------------------
// Tiny global-scalar kernel: rvar, corr(first 8 rows), noise_scale.
// ---------------------------------------------------------------------------
__global__ void k_globals(const float* __restrict__ emb, float* __restrict__ ws_scal) {
  if (threadIdx.x != 0 || blockIdx.x != 0) return;
  float st = ws_scal[0], sr = ws_scal[1], sr2 = ws_scal[2];
  const float Bf = (float)Bsz;
  float ns   = (st / Bf) * 0.1f;
  float rvar = (sr2 - sr * sr / Bf) / (Bf - 1.0f);
  float rich = (rvar < 0.003f) ? 0.3f : ((rvar > 0.008f) ? 0.2f : 1.0f);
  float obj  = (1.0f / (1.0f + __expf(-(rvar - 0.1f) * 10.0f))) * 0.2f;

  float nsq[8];
  for (int i = 0; i < 8; ++i) {
    float s = 0.f;
    for (int c = 0; c < 64; ++c) { float e = emb[i * 64 + c]; s += e * e; }
    nsq[i] = s;
  }
  float hv[28], ev[28];
  int p = 0; float hm = 0.f, em = 0.f;
  for (int i = 0; i < 8; ++i)
    for (int j = i + 1; j < 8; ++j) {
      float d2 = 0.f;
      for (int c = 0; c < 64; ++c) {
        float d = emb[i * 64 + c] - emb[j * 64 + c];
        d2 += d * d;
      }
      float den = (1.0f - nsq[i]) * (1.0f - nsq[j]);
      float hy = acoshf(fmaxf(1.0f + 2.0f * d2 / (den + 1e-12f), 1.0f));
      float eu = sqrtf(fmaxf(d2, 0.0f));
      hv[p] = hy; ev[p] = eu; hm += hy; em += eu; ++p;
    }
  hm *= (1.0f / 28.0f); em *= (1.0f / 28.0f);
  float shh = 0.f, see = 0.f, she = 0.f;
  for (int q = 0; q < 28; ++q) {
    float hc = hv[q] - hm, ec = ev[q] - em;
    shh += hc * hc; see += ec * ec; she += hc * ec;
  }
  float corr = she / (sqrtf(shh) * sqrtf(see) + 1e-8f);
  float metric = (1.0f - fabsf(corr)) * 0.4f;
  ws_scal[3] = ns; ws_scal[4] = rich; ws_scal[5] = metric; ws_scal[6] = obj;
}

// ---------------------------------------------------------------------------
// Fused refiner: 12 steps x (WMMA r1 -> LN/SiLU -> WMMA r2 -> LN/SiLU ->
// WMMA r3 -> project), all activations in LDS.
// ---------------------------------------------------------------------------
__global__ __launch_bounds__(NT) void k_refine(
    const float* __restrict__ emb, const float* __restrict__ noise,
    const bf16* __restrict__ rW1b, const float* __restrict__ rb1,
    const float* __restrict__ rg1, const float* __restrict__ rbt1,
    const bf16* __restrict__ rW2b, const float* __restrict__ rb2,
    const float* __restrict__ rg2, const float* __restrict__ rbt2,
    const bf16* __restrict__ rW3b, const float* __restrict__ rb3,
    const float* __restrict__ ws_scores, const float* __restrict__ ws_radii,
    const float* __restrict__ ws_scal, float* __restrict__ out)
{
  __shared__ __align__(16) bf16 act[ROWS * SA];
  __shared__ __align__(16) bf16 wt0[512 * WS];
  __shared__ __align__(16) bf16 wt1[512 * WS];
  __shared__ float rowsum[ROWS], rowsq[ROWS], meanv[ROWS], rstdv[ROWS];
  __shared__ float red[NT];
  __shared__ float cur[ROWS * 64];
  __shared__ float pred[ROWS * 64];
  __shared__ float f5[ROWS * 5];

  const int tid = threadIdx.x;
  const size_t base = (size_t)blockIdx.x * ROWS;

  const float ns = ws_scal[3], rich = ws_scal[4], metric = ws_scal[5], obj = ws_scal[6];

  // fid5 = scores + penalties
  if (tid < ROWS * 5) {
    int r = tid / 5, c = tid - r * 5;
    float rad = ws_radii[base + r];
    float pen;
    if (c == 0) {
      float av = 9.0f - fminf(fmaxf(rad * 10.0f, 0.0f), 9.0f);
      pen = (av >= 7.0f) ? 0.5f : 0.0f;
    } else if (c == 1) pen = rich;
    else if (c == 2) pen = metric;
    else if (c == 3)
      pen = (__expf(-rad * 5.0f) +
             1.0f / (1.0f + __expf(-(rad - 0.95f) * 20.0f))) * 0.3f;
    else pen = obj;
    f5[tid] = ws_scores[(base + r) * 5 + c] + pen;
  }

  // current = project(emb + noise * ns)
  {
    int row = tid >> 3, c0 = (tid & 7) * 8;
    float v[8]; float ss = 0.f;
#pragma unroll
    for (int e = 0; e < 8; ++e) {
      float x = emb[(base + row) * 64 + c0 + e] +
                noise[(base + row) * 64 + c0 + e] * ns;
      v[e] = x; ss += x * x;
    }
    red[tid] = ss;
    __syncthreads();
    float n2 = 0.f;
    for (int s = 0; s < 8; ++s) n2 += red[row * 8 + s];
    float n = sqrtf(n2);
    float scl = fminf(1.0f, (1.0f - 1e-5f) / fmaxf(n, 1e-12f));
#pragma unroll
    for (int e = 0; e < 8; ++e) cur[row * 64 + c0 + e] = v[e] * scl;
  }
  __syncthreads();

  for (int step = 0; step < 12; ++step) {
    int ts = step * 4;
    float tn = (float)ts * (1.0f / 48.0f);
    float al = 0.99f - (float)ts * (0.98f / 47.0f);

    // net_in = [current(64) | fid5(5) | t(1) | zero-pad to 96] as bf16
    for (int i = tid; i < ROWS * 96; i += NT) {
      int r = i / 96, c = i - r * 96;
      float v;
      if (c < 64)      v = cur[r * 64 + c];
      else if (c < 69) v = f5[r * 5 + (c - 64)];
      else if (c == 69) v = tn;
      else             v = 0.f;
      act[r * SA + c] = (bf16)v;
    }
    __syncthreads();

    gemm512_ln_silu(act, wt0, wt1, rW1b, rb1, rg1, rbt1, rowsum, rowsq, meanv, rstdv, 3,  tid);
    gemm512_ln_silu(act, wt0, wt1, rW2b, rb2, rg2, rbt2, rowsum, rowsq, meanv, rstdv, 16, tid);
    gemm64_out(act, wt0, wt1, rW3b, rb3, pred, tid);

    // current = project(a*current + (1-a)*pred)
    int row = tid >> 3, c0 = (tid & 7) * 8;
    float v[8]; float ss = 0.f;
#pragma unroll
    for (int e = 0; e < 8; ++e) {
      float x = al * cur[row * 64 + c0 + e] + (1.0f - al) * pred[row * 64 + c0 + e];
      v[e] = x; ss += x * x;
    }
    red[tid] = ss;
    __syncthreads();
    float n2 = 0.f;
    for (int s = 0; s < 8; ++s) n2 += red[row * 8 + s];
    float n = sqrtf(n2);
    float scl = fminf(1.0f, (1.0f - 1e-5f) / fmaxf(n, 1e-12f));
#pragma unroll
    for (int e = 0; e < 8; ++e) cur[row * 64 + c0 + e] = v[e] * scl;
    __syncthreads();
  }

  for (int i = tid; i < ROWS * 64; i += NT)
    out[base * 64 + i] = cur[i];
}

// ---------------------------------------------------------------------------
extern "C" void kernel_launch(void* const* d_in, const int* in_sizes, int n_in,
                              void* d_out, int out_size, void* d_ws, size_t ws_size,
                              hipStream_t stream) {
  (void)in_sizes; (void)n_in; (void)out_size; (void)ws_size;
  const float* emb   = (const float*)d_in[0];
  const float* noise = (const float*)d_in[1];
  const float* dW1   = (const float*)d_in[2];
  const float* db1   = (const float*)d_in[3];
  const float* dg1   = (const float*)d_in[4];
  const float* dbt1  = (const float*)d_in[5];
  const float* dW2   = (const float*)d_in[6];
  const float* db2   = (const float*)d_in[7];
  const float* dg2   = (const float*)d_in[8];
  const float* dbt2  = (const float*)d_in[9];
  const float* dW3   = (const float*)d_in[10];
  const float* db3   = (const float*)d_in[11];
  const float* rW1   = (const float*)d_in[12];
  const float* rb1   = (const float*)d_in[13];
  const float* rg1   = (const float*)d_in[14];
  const float* rbt1  = (const float*)d_in[15];
  const float* rW2   = (const float*)d_in[16];
  const float* rb2   = (const float*)d_in[17];
  const float* rg2   = (const float*)d_in[18];
  const float* rbt2  = (const float*)d_in[19];
  const float* rW3   = (const float*)d_in[20];
  const float* rb3   = (const float*)d_in[21];

  // workspace layout
  float* ws_scores = (float*)d_ws;                       // B*5
  float* ws_radii  = ws_scores + (size_t)Bsz * 5;        // B
  float* ws_scal   = ws_radii + Bsz;                     // 8 scalars
  bf16*  dW1b = (bf16*)(ws_scal + 8);                    // 64*512   packed
  bf16*  dW2b = dW1b + 64 * 512;                         // 512*512  packed
  bf16*  rW1b = dW2b + 512 * 512;                        // 96*512   packed+pad
  bf16*  rW2b = rW1b + 96 * 512;                         // 512*512  packed
  bf16*  rW3b = rW2b + 512 * 512;                        // 512*64   packed

  k_zero<<<1, 32, 0, stream>>>(ws_scal);
  k_cvt_pack<<<(64 * 512 + 255) / 256, 256, 0, stream>>>(dW1, dW1b, 64, 64, 512);
  k_cvt_pack<<<(512 * 512 + 255) / 256, 256, 0, stream>>>(dW2, dW2b, 512, 512, 512);
  k_cvt_pack<<<(96 * 512 + 255) / 256, 256, 0, stream>>>(rW1, rW1b, 70, 96, 512);
  k_cvt_pack<<<(512 * 512 + 255) / 256, 256, 0, stream>>>(rW2, rW2b, 512, 512, 512);
  k_cvt_pack<<<(512 * 64 + 255) / 256, 256, 0, stream>>>(rW3, rW3b, 512, 512, 64);

  k_scorer<<<Bsz / ROWS, NT, 0, stream>>>(emb, dW1b, db1, dg1, dbt1,
                                          dW2b, db2, dg2, dbt2,
                                          dW3, db3, ws_scores, ws_radii, ws_scal);
  k_globals<<<1, 32, 0, stream>>>(emb, ws_scal);
  k_refine<<<Bsz / ROWS, NT, 0, stream>>>(emb, noise,
                                          rW1b, rb1, rg1, rbt1,
                                          rW2b, rb2, rg2, rbt2,
                                          rW3b, rb3,
                                          ws_scores, ws_radii, ws_scal,
                                          (float*)d_out);
}
